// UNO_1D_77704548319515
// MI455X (gfx1250) — matrix operand order repaired and checked by reading, not compiled, not run on hardware
//
#include <hip/hip_runtime.h>
#include <math.h>

// ---------------------------------------------------------------------------
// CDNA5 (gfx1250) UNO-1D forward. wave32. WMMA f32<-f16 16x16x32 for all
// channel-mixing GEMMs; VALU for small DFT/irDFT/norm/gelu epilogues.
// fc1 GEMM stages tiles with GLOBAL_LOAD_ASYNC_TO_LDS_B128 via inline asm
// (ASYNCcnt path); gather-style stagings use global_prefetch hints.
// ---------------------------------------------------------------------------

typedef _Float16 half8   __attribute__((ext_vector_type(8)));
typedef _Float16 half16  __attribute__((ext_vector_type(16)));
typedef float    float4v __attribute__((ext_vector_type(4)));
typedef float    float8  __attribute__((ext_vector_type(8)));

#define WMMA(a, b, c) \
  __builtin_amdgcn_wmma_f32_16x16x32_f16(false, (a), false, (b), (short)0, (c), false, false)

// Async copy 16B global -> LDS. Generic LDS pointers carry the LDS byte
// offset in their low 32 bits (ISA 10.2 aperture rules), which is exactly the
// VDST operand of GLOBAL_LOAD_ASYNC_TO_LDS_B128. Inline asm bypasses the
// builtin's address-space-qualified prototype (toolchain-dependent).
__device__ __forceinline__ void async_b128(const void* gsrc, void* ldst) {
  const unsigned lds32 = (unsigned)(unsigned long long)ldst;
  asm volatile("global_load_async_to_lds_b128 %0, %1, off"
               :
               : "v"(lds32), "v"(gsrc)
               : "memory");
}

__device__ __forceinline__ void wait_async0() {
#if defined(__has_builtin) && __has_builtin(__builtin_amdgcn_s_wait_asynccnt)
  __builtin_amdgcn_s_wait_asynccnt(0);
#else
  asm volatile("s_wait_asynccnt 0x0" ::: "memory");
#endif
}

__device__ __forceinline__ float gelu_f(float v) {
  return 0.5f * v * (1.0f + erff(v * 0.70710678118654752f));
}

// A fragment (16x32 f16, ISA 7.12.2): lane L<16 -> row M=L, K in {0..7,16..23};
// lane>=16 -> row M=L-16, K in {8..15,24..31}. LDS tile row stride = 32 halves.
__device__ __forceinline__ half16 frag_a(const _Float16* As, int lane) {
  const int row = lane & 15;
  const int kb  = (lane >> 4) << 3;
  half8 lo = *(const half8*)(As + row * 32 + kb);
  half8 hi = *(const half8*)(As + row * 32 + kb + 16);
  return __builtin_shufflevector(lo, hi, 0, 1, 2, 3, 4, 5, 6, 7, 8, 9, 10, 11, 12, 13, 14, 15);
}

// B fragment (32x16 f16): lane L<16 -> col N=L, K in {0..7,16..23}; lane>=16 -> K in {8..15,24..31}.
// B tile stored TRANSPOSED in LDS: Bs[n*32 + k] so per-lane reads are contiguous.
__device__ __forceinline__ half16 frag_b(const _Float16* Bs, int lane) {
  const int col = lane & 15;
  const int kb  = (lane >> 4) << 3;
  half8 lo = *(const half8*)(Bs + col * 32 + kb);
  half8 hi = *(const half8*)(Bs + col * 32 + kb + 16);
  return __builtin_shufflevector(lo, hi, 0, 1, 2, 3, 4, 5, 6, 7, 8, 9, 10, 11, 12, 13, 14, 15);
}

// Same fragments but from f32 LDS tiles (filled by async-to-LDS copies);
// f32 -> f16 conversion fused into fragment assembly.
__device__ __forceinline__ half16 frag_a32(const float* As, int lane) {
  const int row = lane & 15;
  const int kb  = (lane >> 4) << 3;
  float4v q0 = *(const float4v*)(As + row * 32 + kb);
  float4v q1 = *(const float4v*)(As + row * 32 + kb + 4);
  float4v q2 = *(const float4v*)(As + row * 32 + kb + 16);
  float4v q3 = *(const float4v*)(As + row * 32 + kb + 20);
  float8 lo = __builtin_shufflevector(q0, q1, 0, 1, 2, 3, 4, 5, 6, 7);
  float8 hi = __builtin_shufflevector(q2, q3, 0, 1, 2, 3, 4, 5, 6, 7);
  half8 l = __builtin_convertvector(lo, half8);
  half8 h = __builtin_convertvector(hi, half8);
  return __builtin_shufflevector(l, h, 0, 1, 2, 3, 4, 5, 6, 7, 8, 9, 10, 11, 12, 13, 14, 15);
}

__device__ __forceinline__ half16 frag_b32(const float* Bs, int lane) {
  const int col = lane & 15;
  const int kb  = (lane >> 4) << 3;
  float4v q0 = *(const float4v*)(Bs + col * 32 + kb);
  float4v q1 = *(const float4v*)(Bs + col * 32 + kb + 4);
  float4v q2 = *(const float4v*)(Bs + col * 32 + kb + 16);
  float4v q3 = *(const float4v*)(Bs + col * 32 + kb + 20);
  float8 lo = __builtin_shufflevector(q0, q1, 0, 1, 2, 3, 4, 5, 6, 7);
  float8 hi = __builtin_shufflevector(q2, q3, 0, 1, 2, 3, 4, 5, 6, 7);
  half8 l = __builtin_convertvector(lo, half8);
  half8 h = __builtin_convertvector(hi, half8);
  return __builtin_shufflevector(l, h, 0, 1, 2, 3, 4, 5, 6, 7, 8, 9, 10, 11, 12, 13, 14, 15);
}

// ------------------------- 1) token MLP: fc + fc0 --------------------------
// x (B,S,16) -> h (B,128,64) channel-major.  B=1024, S=64.
__global__ __launch_bounds__(128) void k_embed(
    const float* __restrict__ x, const float* __restrict__ fcw,
    const float* __restrict__ fcb, const float* __restrict__ f0w,
    const float* __restrict__ f0b, float* __restrict__ h) {
  const int t = blockIdx.x * 128 + threadIdx.x;
  if (t >= 1024 * 64) return;
  const int b = t >> 6, s = t & 63;
  float in[17];
  const float* xp = x + (size_t)t * 16;
#pragma unroll
  for (int i = 0; i < 16; ++i) in[i] = xp[i];
  const float gx = 6.28318530717958647f * (float)s / 63.0f;
  in[16] = sinf(gx) + cosf(gx);
  float h1[64];
  for (int j = 0; j < 64; ++j) {
    float acc = fcb[j];
#pragma unroll
    for (int i = 0; i < 17; ++i) acc += fcw[j * 17 + i] * in[i];
    h1[j] = gelu_f(acc);
  }
  for (int c = 0; c < 128; ++c) {
    float acc = f0b[c];
#pragma unroll 8
    for (int j = 0; j < 64; ++j) acc += f0w[c * 64 + j] * h1[j];
    h[((size_t)b * 128 + c) * 64 + s] = gelu_f(acc);
  }
}

// ------------------------- 2) forward rDFT (norm='forward') ----------------
// xin (rows=B*Cin, L) -> Xr/Xi (rows, modes)
__global__ __launch_bounds__(128) void k_dft(
    const float* __restrict__ xin, float* __restrict__ Xr, float* __restrict__ Xi,
    int rows, int L, int modes) {
  const int t = blockIdx.x * 128 + threadIdx.x;
  if (t >= rows * modes) return;
  const int row = t / modes, k = t - row * modes;
  const float* xp = xin + (size_t)row * L;
  const float w = 6.28318530717958647f * (float)k / (float)L;
  float cr = 0.0f, ci = 0.0f;
  for (int l = 0; l < L; ++l) {
    float sn, cs;
    sincosf(w * (float)l, &sn, &cs);
    const float v = xp[l];
    cr += v * cs;
    ci -= v * sn;
  }
  const float inv = 1.0f / (float)L;
  Xr[t] = cr * inv;
  Xi[t] = ci * inv;
}

// ------------------------- 3) per-mode complex GEMM (WMMA) -----------------
// Or + i*Oi = (Xr + i*Xi)(b,:) @ (Wr + i*Wi)(:,:,k).  M=B=1024, K=Cin, N=Cout.
// 4 waves/WG -> 64x16 C tile; f16 WMMA has no A-neg, so negate the Ai fragment.
__global__ __launch_bounds__(128) void k_modemix(
    const float* __restrict__ Xr, const float* __restrict__ Xi,
    const float* __restrict__ wr, const float* __restrict__ wi,
    float* __restrict__ Orr, float* __restrict__ Oii,
    int Cin, int Cout, int modes) {
  const int mode = blockIdx.z;
  const int n0 = blockIdx.x * 16;
  const int m0 = blockIdx.y * 64;
  __shared__ _Float16 Ar[64 * 32] __attribute__((aligned(16)));
  __shared__ _Float16 Ai[64 * 32] __attribute__((aligned(16)));
  __shared__ _Float16 Br[16 * 32] __attribute__((aligned(16)));
  __shared__ _Float16 Bi[16 * 32] __attribute__((aligned(16)));
  const int tid = threadIdx.x, wid = tid >> 5, lane = tid & 31;
  float8 accr = {}, acci = {};
  for (int k0 = 0; k0 < Cin; k0 += 32) {
    if (k0 + 32 < Cin) {  // prefetch next K-tile of the A operand
      const size_t pidx = ((size_t)(m0 + (tid >> 1)) * Cin + (k0 + 32 + ((tid & 1) << 4))) *
                              (size_t)modes + mode;
      __builtin_prefetch(Xr + pidx, 0, 1);
      __builtin_prefetch(Xi + pidx, 0, 1);
    }
    for (int e = tid; e < 64 * 32; e += 128) {
      const int r = e >> 5, kk = e & 31;
      const size_t idx = ((size_t)(m0 + r) * Cin + (k0 + kk)) * (size_t)modes + mode;
      Ar[e] = (_Float16)Xr[idx];
      Ai[e] = (_Float16)Xi[idx];
    }
    for (int e = tid; e < 16 * 32; e += 128) {
      const int o = e >> 5, i = e & 31;  // Bs[o*32 + i] = W[k0+i][n0+o]
      const size_t idx = ((size_t)(k0 + i) * Cout + (n0 + o)) * (size_t)modes + mode;
      Br[e] = (_Float16)wr[idx];
      Bi[e] = (_Float16)wi[idx];
    }
    __syncthreads();
    const half16 ar = frag_a(Ar + wid * 16 * 32, lane);
    const half16 ai = frag_a(Ai + wid * 16 * 32, lane);
    const half16 br = frag_b(Br, lane);
    const half16 bi = frag_b(Bi, lane);
    half16 z = {};
    const half16 nai = z - ai;
    accr = WMMA(ar, br, accr);
    accr = WMMA(nai, bi, accr);   // - Xi*Wi
    acci = WMMA(ar, bi, acci);
    acci = WMMA(ai, br, acci);
    __syncthreads();
  }
  const int mbase = m0 + wid * 16 + ((lane < 16) ? 0 : 8);
  const int col = n0 + (lane & 15);
#pragma unroll
  for (int r = 0; r < 8; ++r) {
    const size_t idx = ((size_t)(mbase + r) * Cout + col) * (size_t)modes + mode;
    Orr[idx] = accr[r];
    Oii[idx] = acci[r];
  }
}

// ------------------------- 4) 1x1 conv skip (WMMA, batched) ----------------
// x2[b] (Cout,L) = cw (Cout,Cin) @ xin[b] (Cin,L); bias added in finalize.
__global__ __launch_bounds__(128) void k_conv1x1(
    const float* __restrict__ xin, const float* __restrict__ cw,
    float* __restrict__ x2, int Cin, int Cout, int L) {
  const int b = blockIdx.z;
  const int m0 = blockIdx.y * 64;
  const int n0 = blockIdx.x * 16;
  __shared__ _Float16 As[64 * 32] __attribute__((aligned(16)));
  __shared__ _Float16 Bs[16 * 32] __attribute__((aligned(16)));
  const int tid = threadIdx.x, wid = tid >> 5, lane = tid & 31;
  float8 acc = {};
  for (int k0 = 0; k0 < Cin; k0 += 32) {
    if (k0 + 32 < Cin) {  // prefetch next K-tile of weights and activations
      __builtin_prefetch(cw + (size_t)(m0 + (tid >> 1)) * Cin + k0 + 32 + ((tid & 1) << 4), 0, 1);
      __builtin_prefetch(xin + ((size_t)b * Cin + (k0 + 32 + (tid & 31))) * L + n0, 0, 1);
    }
    for (int e = tid; e < 64 * 32; e += 128) {
      const int r = e >> 5, kk = e & 31;
      As[e] = (_Float16)cw[(size_t)(m0 + r) * Cin + k0 + kk];
    }
    for (int e = tid; e < 16 * 32; e += 128) {
      const int o = e >> 5, i = e & 31;
      Bs[e] = (_Float16)xin[((size_t)b * Cin + (k0 + i)) * L + (n0 + o)];
    }
    __syncthreads();
    const half16 a = frag_a(As + wid * 16 * 32, lane);
    const half16 bf = frag_b(Bs, lane);
    acc = WMMA(a, bf, acc);
    __syncthreads();
  }
  const int mbase = m0 + wid * 16 + ((lane < 16) ? 0 : 8);
  const int col = n0 + (lane & 15);
#pragma unroll
  for (int r = 0; r < 8; ++r)
    x2[((size_t)b * Cout + (mbase + r)) * L + col] = acc[r];
}

// ------------------------- 5) irDFT + interp + IN + GELU -------------------
// blockDim = (dim1, 4); one (b, o) row per (blockIdx.y, blockIdx.x*4+ty).
__global__ void k_finalize(
    const float* __restrict__ Orr, const float* __restrict__ Oii,
    const float* __restrict__ x2, const float* __restrict__ cb,
    const float* __restrict__ gam, const float* __restrict__ bet,
    float* __restrict__ dst, int Cout, int modes, int Lin, int dim1,
    int dstC, int choff) {
  __shared__ float cosb[22 * 64], sinb[22 * 64];
  __shared__ float rs[4][64], rq[4][64];
  __shared__ float mv[4][2];
  const int tx = threadIdx.x, ty = threadIdx.y;
  const int nth = blockDim.x * blockDim.y;
  const int tid = ty * blockDim.x + tx;
  for (int e = tid; e < modes * dim1; e += nth) {
    const int k = e / dim1, tt = e - k * dim1;
    float sn, cs;
    sincosf(6.28318530717958647f * (float)(k * tt) / (float)dim1, &sn, &cs);
    cosb[e] = cs;
    sinb[e] = sn;
  }
  __syncthreads();
  const int b = blockIdx.y;
  const int o = blockIdx.x * 4 + ty;
  const float* Orp = Orr + ((size_t)b * Cout + o) * modes;
  const float* Oip = Oii + ((size_t)b * Cout + o) * modes;
  // irfft (norm='forward' inverse has no 1/n); modes < dim1/2+1 so weight=2 for k>0
  float v = Orp[0] * cosb[tx] - Oip[0] * sinb[tx];
  for (int k = 1; k < modes; ++k)
    v += 2.0f * (Orp[k] * cosb[k * dim1 + tx] - Oip[k] * sinb[k * dim1 + tx]);
  // linear interp (align_corners=True) of conv skip + bias
  const float* xp = x2 + ((size_t)b * Cout + o) * Lin;
  const float pos = (float)tx * (float)(Lin - 1) / (float)(dim1 - 1);
  int lo = (int)floorf(pos);
  lo = lo < 0 ? 0 : (lo > Lin - 2 ? Lin - 2 : lo);
  const float fr = pos - (float)lo;
  v += xp[lo] * (1.0f - fr) + xp[lo + 1] * fr + cb[o];
  rs[ty][tx] = v;
  rq[ty][tx] = v * v;
  __syncthreads();
  if (tx == 0) {
    float s = 0.0f, q = 0.0f;
    for (int i = 0; i < blockDim.x; ++i) { s += rs[ty][i]; q += rq[ty][i]; }
    const float mu = s / (float)blockDim.x;
    mv[ty][0] = mu;
    mv[ty][1] = rsqrtf(q / (float)blockDim.x - mu * mu + 1e-5f);
  }
  __syncthreads();
  const float outv = (v - mv[ty][0]) * mv[ty][1] * gam[o] + bet[o];
  dst[((size_t)b * dstC + choff + o) * dim1 + tx] = gelu_f(outv);
}

// ------------------------- 6) channel-offset copy (U-Net concats) ----------
__global__ __launch_bounds__(256) void k_copych(
    const float* __restrict__ src, float* __restrict__ dst,
    int C, int L, int dstC, int choff) {
  const long long t = (long long)blockIdx.x * 256 + threadIdx.x;
  const long long total = 1024LL * C * L;
  if (t >= total) return;
  const int l = (int)(t % L);
  const long long rc = t / L;
  const int c = (int)(rc % C);
  const int b = (int)(rc / C);
  dst[((size_t)b * dstC + choff + c) * L + l] = src[t];
}

// ------------------------- 7) (B,256,64) -> tokens (B*64, 256) -------------
__global__ __launch_bounds__(256) void k_transpose(
    const float* __restrict__ src, float* __restrict__ dst) {
  const long long t = (long long)blockIdx.x * 256 + threadIdx.x;
  if (t >= 1024LL * 256 * 64) return;
  const int s = (int)(t & 63);
  const long long rc = t >> 6;
  const int c = (int)(rc & 255);
  const int b = (int)(rc >> 8);
  dst[((size_t)b * 64 + s) * 256 + c] = src[t];
}

// ------------------------- 8) fc1 GEMM 65536x256x512 + bias + GELU ---------
// A and B tiles are contiguous 128B rows -> stage raw f32 tiles via
// GLOBAL_LOAD_ASYNC_TO_LDS_B128 (ASYNCcnt), convert f32->f16 in fragment load.
__global__ __launch_bounds__(128) void k_gemm_fc1(
    const float* __restrict__ tok, const float* __restrict__ w,
    const float* __restrict__ bias, float* __restrict__ out) {
  const int m0 = blockIdx.y * 64;
  const int n0 = blockIdx.x * 16;
  __shared__ float Af[64 * 32] __attribute__((aligned(16)));  // 8 KB
  __shared__ float Bf[16 * 32] __attribute__((aligned(16)));  // 2 KB
  const int tid = threadIdx.x, wid = tid >> 5, lane = tid & 31;
  float8 acc = {};
  for (int k0 = 0; k0 < 256; k0 += 32) {
    // A tile: 64 rows x 32 f32 = 512 x 16B chunks, 4 rounds of 128 lanes.
#pragma unroll
    for (int it = 0; it < 4; ++it) {
      const int e = tid + it * 128;
      const int r = e >> 3, ch = (e & 7) << 2;
      async_b128(tok + (size_t)(m0 + r) * 256 + k0 + ch, &Af[r * 32 + ch]);
    }
    // B tile: 16 rows x 32 f32 = 128 x 16B chunks, 1 round.
    {
      const int r = tid >> 3, ch = (tid & 7) << 2;
      async_b128(w + (size_t)(n0 + r) * 256 + k0 + ch, &Bf[r * 32 + ch]);
    }
    wait_async0();
    __syncthreads();
    const half16 a = frag_a32(Af + wid * 16 * 32, lane);
    const half16 bf = frag_b32(Bf, lane);
    acc = WMMA(a, bf, acc);
    __syncthreads();
  }
  const int mbase = m0 + wid * 16 + ((lane < 16) ? 0 : 8);
  const int col = n0 + (lane & 15);
#pragma unroll
  for (int r = 0; r < 8; ++r)
    out[(size_t)(mbase + r) * 512 + col] = gelu_f(acc[r] + bias[col]);
}

// ------------------------- 9) fc2: 512 -> 1 dot ----------------------------
__global__ __launch_bounds__(256) void k_fc2(
    const float* __restrict__ h1, const float* __restrict__ w,
    const float* __restrict__ b, float* __restrict__ out) {
  const int t = blockIdx.x * 256 + threadIdx.x;
  if (t >= 65536) return;
  const float* hp = h1 + (size_t)t * 512;
  float acc = b[0];
#pragma unroll 8
  for (int k = 0; k < 512; ++k) acc += hp[k] * w[k];
  out[t] = acc;
}

// ---------------------------------------------------------------------------
// Host orchestration
// ---------------------------------------------------------------------------
extern "C" void kernel_launch(void* const* d_in, const int* in_sizes, int n_in,
                              void* d_out, int out_size, void* d_ws, size_t ws_size,
                              hipStream_t stream) {
  (void)in_sizes; (void)n_in; (void)out_size; (void)ws_size;
  const float* x    = (const float*)d_in[0];
  const float* fcw  = (const float*)d_in[1];
  const float* fcb  = (const float*)d_in[2];
  const float* f0w  = (const float*)d_in[3];
  const float* f0b  = (const float*)d_in[4];
  const float* fc1w = (const float*)d_in[47];
  const float* fc1b = (const float*)d_in[48];
  const float* fc2w = (const float*)d_in[49];
  const float* fc2b = (const float*)d_in[50];
  float* ws = (float*)d_ws;
  float* out = (float*)d_out;

  // workspace layout (float offsets)
  const size_t OFF_H  = 0;                         // h      (B,128,64)
  const size_t OFF_C0 = 8388608;                   // x_c0   (B,192,48)
  const size_t OFF_C1 = 17825792;                  // x_c1   (B,384,32)
  const size_t OFF_T0 = 30408704;                  // ping   (covers B,768,32)
  const size_t OFF_T1 = 55574528;                  // pong   (covers B,768,32)
  const size_t OFF_XR = 80740352;                  // Xr     (<= B*768*14)
  const size_t OFF_XI = 91750400;                  // Xi
  const size_t OFF_OR = 102760448;                 // Or     (<= B*384*14)
  const size_t OFF_OI = 108265472;                 // Oi
  const size_t OFF_X2 = 113770496;                 // conv   (<= B*768*32)
  const size_t OFF_TOK = OFF_XR;                   // tokens reuse DFT scratch
  const size_t OFF_H1  = OFF_OR;                   // fc1 out reuses Or..X2

  k_embed<<<512, 128, 0, stream>>>(x, fcw, fcb, f0w, f0b, ws + OFF_H);

  const int Cin[7]   = {128, 192, 384, 768, 768, 768, 384};
  const int Cout[7]  = {192, 384, 768, 768, 384, 192, 128};
  const int Modes[7] = {22, 14, 6, 6, 6, 14, 22};
  const int Lin[7]   = {64, 48, 32, 16, 16, 32, 48};
  const int Dim1[7]  = {48, 32, 16, 16, 32, 48, 64};
  const size_t InOff[7]  = {OFF_H, OFF_C0, OFF_C1, OFF_T0, OFF_T1, OFF_T0, OFF_T1};
  const size_t OutOff[7] = {OFF_C0, OFF_C1, OFF_T0, OFF_T1, OFF_T0, OFF_T1, OFF_T0};
  const int DstC[7]  = {192, 384, 768, 768, 768, 384, 256};
  // block 4 writes channels [0,384) of cat5 in T0; x_c1 copied to [384,768).
  // block 5 writes [0,192) of cat6 in T1; x_c0 copied to [192,384).
  // block 6 writes [0,128) of final cat in T0; h copied to [128,256).

  for (int i = 0; i < 7; ++i) {
    const float* wr = (const float*)d_in[5 + 6 * i + 0];
    const float* wi = (const float*)d_in[5 + 6 * i + 1];
    const float* cw = (const float*)d_in[5 + 6 * i + 2];
    const float* cb = (const float*)d_in[5 + 6 * i + 3];
    const float* gg = (const float*)d_in[5 + 6 * i + 4];
    const float* bb = (const float*)d_in[5 + 6 * i + 5];
    const int ci = Cin[i], co = Cout[i], m = Modes[i], li = Lin[i], d1 = Dim1[i];
    const int rows = 1024 * ci;

    k_dft<<<(rows * m + 127) / 128, 128, 0, stream>>>(
        ws + InOff[i], ws + OFF_XR, ws + OFF_XI, rows, li, m);
    k_modemix<<<dim3(co / 16, 16, m), 128, 0, stream>>>(
        ws + OFF_XR, ws + OFF_XI, wr, wi, ws + OFF_OR, ws + OFF_OI, ci, co, m);
    k_conv1x1<<<dim3(li / 16, co / 64, 1024), 128, 0, stream>>>(
        ws + InOff[i], cw, ws + OFF_X2, ci, co, li);
    k_finalize<<<dim3(co / 4, 1024), dim3(d1, 4), 0, stream>>>(
        ws + OFF_OR, ws + OFF_OI, ws + OFF_X2, cb, gg, bb, ws + OutOff[i],
        co, m, li, d1, DstC[i], 0);

    if (i == 4) {  // cat5: append x_c1 after block4 output in T0
      const long long tot = 1024LL * 384 * 32;
      k_copych<<<(unsigned)((tot + 255) / 256), 256, 0, stream>>>(
          ws + OFF_C1, ws + OFF_T0, 384, 32, 768, 384);
    } else if (i == 5) {  // cat6: append x_c0 after block5 output in T1
      const long long tot = 1024LL * 192 * 48;
      k_copych<<<(unsigned)((tot + 255) / 256), 256, 0, stream>>>(
          ws + OFF_C0, ws + OFF_T1, 192, 48, 384, 192);
    } else if (i == 6) {  // final cat: append h after block6 output in T0
      const long long tot = 1024LL * 128 * 64;
      k_copych<<<(unsigned)((tot + 255) / 256), 256, 0, stream>>>(
          ws + OFF_H, ws + OFF_T0, 128, 64, 256, 128);
    }
  }

  k_transpose<<<65536, 256, 0, stream>>>(ws + OFF_T0, ws + OFF_TOK);
  k_gemm_fc1<<<dim3(32, 1024), 128, 0, stream>>>(ws + OFF_TOK, fc1w, fc1b, ws + OFF_H1);
  k_fc2<<<256, 256, 0, stream>>>(ws + OFF_H1, fc2w, fc2b, out);
}
